// ROEN_32066225832583
// MI455X (gfx1250) — matrix operand order repaired and checked by reading, not compiled, not run on hardware
//
#include <hip/hip_runtime.h>

// ---------------------------------------------------------------------------
// MI455X (gfx1250, wave32) fused graph-transformer pipeline.
// Dense layers: v_wmma_f32_16x16x32_bf16 (bf16 in, f32 accum), weights
// pre-packed once into WMMA B-operand register layout (one v16bf load/operand).
// Classifier feature gather uses GLOBAL_LOAD_ASYNC_TO_LDS_B128 + s_wait_asynccnt.
// All row counts are multiples of 16 -> branch-free tiles, EXEC always full.
// ---------------------------------------------------------------------------

#define WPB 4  // waves per 128-thread block

typedef __attribute__((ext_vector_type(16))) __bf16 v16bf;
typedef __attribute__((ext_vector_type(8)))  float  v8f;

// Native f32->bf16 (RNE); backend emits packed v_cvt ops.
static __device__ __forceinline__ __bf16 f2bf(float f) { return (__bf16)f; }

// A operand (16x32 bf16): caller passes this lane's row base (row = lane&15).
// ISA 7.12.2: lane half h holds K = {h*8+0..7, 16+h*8+0..7} within the chunk.
static __device__ __forceinline__ v16bf packA(const float* __restrict__ row, int kc, int half) {
  v16bf a;
#pragma unroll
  for (int j = 0; j < 8; ++j) a[j] = f2bf(row[kc + half * 8 + j]);
#pragma unroll
  for (int j = 0; j < 8; ++j) a[8 + j] = f2bf(row[kc + 16 + half * 8 + j]);
  return a;
}

// Pre-packed B operand: Wp laid out as [(nt*KC + kc)*32 + lane] x v16bf.
static __device__ __forceinline__ v16bf loadBp(const __bf16* __restrict__ Wp,
                                               int KC, int kc, int nt, int lane) {
  const v16bf* p = (const v16bf*)Wp;
  return p[((size_t)nt * KC + kc) * 32 + lane];
}

static __device__ __forceinline__ v8f wmma_bf16(v16bf a, v16bf b, v8f c) {
  return __builtin_amdgcn_wmma_f32_16x16x32_bf16(false, a, false, b, (short)0, c, false, false);
}

// Async gather: copy 16 aligned bytes global -> LDS (ASYNCcnt-tracked).
static __device__ __forceinline__ void async_g2l_b128(void* lds_ptr, const void* gptr) {
  unsigned ldso = (unsigned)(uintptr_t)lds_ptr;  // low 32 bits = LDS byte offset
  asm volatile("global_load_async_to_lds_b128 %0, %1, off"
               :: "v"(ldso), "v"(gptr) : "memory");
}
static __device__ __forceinline__ void wait_async0() {
  asm volatile("s_wait_asynccnt 0x0" ::: "memory");
}

// ---------------- ordered-uint mapping for float atomicMax ----------------
static __device__ __forceinline__ unsigned fmapu(float f) {
  unsigned u = __builtin_bit_cast(unsigned, f);
  return (u & 0x80000000u) ? ~u : (u | 0x80000000u);
}
static __device__ __forceinline__ float fumap(unsigned u) {
  u = (u & 0x80000000u) ? (u & 0x7FFFFFFFu) : ~u;
  return __builtin_bit_cast(float, u);
}

// ===========================================================================
// Weight pre-pack: f32 row-major W[K x Nout] -> bf16 WMMA B layout.
// One thread per (nt, kc, lane) writes 16 bf16 (lane's B register payload).
// ===========================================================================
__global__ void packW_kernel(const float* __restrict__ W, int K, int Nout,
                             __bf16* __restrict__ out) {
  int KC = K >> 5, NT = (Nout + 15) >> 4;
  int idx = blockIdx.x * blockDim.x + threadIdx.x;
  if (idx >= KC * NT * 32) return;
  int lane = idx & 31, rest = idx >> 5;
  int kc = rest % KC, nt = rest / KC;
  int half = lane >> 4, lr = lane & 15;
  int ncol = nt * 16 + lr; if (ncol >= Nout) ncol = 0;  // pad cols (clf2 Nout=8)
  int kb = kc * 32 + half * 8;
#pragma unroll
  for (int j = 0; j < 16; ++j)
    out[(size_t)idx * 16 + j] = f2bf(W[(size_t)(kb + (j & 7) + (j >> 3) * 16) * Nout + ncol]);
}

// ===========================================================================
// Fused 2-layer MLP: Y = relu(relu(X@W1+b1)@W2+b2).  K1 in {32,64}, 32->32.
// rows must be a multiple of 16 (holds for all call sites).
// ===========================================================================
template <int K1>
__global__ void mlp2_kernel(int rows, const float* __restrict__ X,
                            const __bf16* __restrict__ W1p, const float* __restrict__ b1,
                            const __bf16* __restrict__ W2p, const float* __restrict__ b2,
                            float* __restrict__ Y) {
  __shared__ float hid[WPB][16][32];
  constexpr int KC1 = K1 / 32;
  int lane = threadIdx.x & 31, wid = threadIdx.x >> 5;
  int half = lane >> 4, lr = lane & 15;
  int tiles = rows >> 4;
  __builtin_prefetch(W1p, 0, 1);
  __builtin_prefetch(W2p, 0, 1);
  for (int tile = blockIdx.x * WPB + wid; tile < tiles; tile += gridDim.x * WPB) {
    const float* xr = X + (size_t)(tile * 16 + lr) * K1;
    v16bf a[KC1];
#pragma unroll
    for (int c = 0; c < KC1; ++c) a[c] = packA(xr, c * 32, half);
#pragma unroll
    for (int nt = 0; nt < 2; ++nt) {
      v8f acc = {};
#pragma unroll
      for (int c = 0; c < KC1; ++c)
        acc = wmma_bf16(a[c], loadBp(W1p, KC1, c, nt, lane), acc);
      int col = nt * 16 + lr; float bb = b1[col];
#pragma unroll
      for (int i2 = 0; i2 < 8; ++i2) {
        float vv = acc[i2] + bb;
        hid[wid][i2 + half * 8][col] = vv > 0.f ? vv : 0.f;
      }
    }
    v16bf a2 = packA(&hid[wid][lr][0], 0, half);
#pragma unroll
    for (int nt = 0; nt < 2; ++nt) {
      v8f acc = {};
      acc = wmma_bf16(a2, loadBp(W2p, 1, 0, nt, lane), acc);
      int col = nt * 16 + lr; float bb = b2[col];
      float* yp = Y + ((size_t)(tile * 16 + half * 8)) * 32 + col;
#pragma unroll
      for (int i2 = 0; i2 < 8; ++i2) {
        float vv = acc[i2] + bb;
        yp[(size_t)i2 * 32] = vv > 0.f ? vv : 0.f;
      }
    }
  }
}

// ===========================================================================
// Plain 32->32 GEMM (+bias):  Y = X@W + b
// ===========================================================================
__global__ void gemm32_kernel(int rows, const float* __restrict__ X,
                              const __bf16* __restrict__ Wp, const float* __restrict__ bias,
                              float* __restrict__ Y) {
  int lane = threadIdx.x & 31, wid = threadIdx.x >> 5;
  int half = lane >> 4, lr = lane & 15;
  int tiles = rows >> 4;
  __builtin_prefetch(Wp, 0, 1);
  for (int tile = blockIdx.x * WPB + wid; tile < tiles; tile += gridDim.x * WPB) {
    v16bf a = packA(X + (size_t)(tile * 16 + lr) * 32, 0, half);
#pragma unroll
    for (int nt = 0; nt < 2; ++nt) {
      v8f acc = {};
      acc = wmma_bf16(a, loadBp(Wp, 1, 0, nt, lane), acc);
      int col = nt * 16 + lr; float bb = bias[col];
      float* yp = Y + ((size_t)(tile * 16 + half * 8)) * 32 + col;
#pragma unroll
      for (int i2 = 0; i2 < 8; ++i2) yp[(size_t)i2 * 32] = acc[i2] + bb;
    }
  }
}

// ===========================================================================
// Y = LayerNorm(resid + X@W + b)  (rows x 32)
// ===========================================================================
__global__ void oln_kernel(int rows, const float* __restrict__ X,
                           const __bf16* __restrict__ Wp, const float* __restrict__ bias,
                           const float* __restrict__ resid,
                           const float* __restrict__ gamma, const float* __restrict__ beta,
                           float* __restrict__ Y) {
  __shared__ float tileb[WPB][16][32];
  int lane = threadIdx.x & 31, wid = threadIdx.x >> 5;
  int half = lane >> 4, lr = lane & 15;
  int tiles = rows >> 4;
  __builtin_prefetch(Wp, 0, 1);
  for (int tile = blockIdx.x * WPB + wid; tile < tiles; tile += gridDim.x * WPB) {
    v16bf a = packA(X + (size_t)(tile * 16 + lr) * 32, 0, half);
#pragma unroll
    for (int nt = 0; nt < 2; ++nt) {
      v8f acc = {};
      acc = wmma_bf16(a, loadBp(Wp, 1, 0, nt, lane), acc);
      int col = nt * 16 + lr; float bb = bias[col];
#pragma unroll
      for (int i2 = 0; i2 < 8; ++i2) tileb[wid][i2 + half * 8][col] = acc[i2] + bb;
    }
    if (lane < 16) {
      int gr = tile * 16 + lane;
      float vals[32]; float m = 0.f;
#pragma unroll
      for (int c = 0; c < 32; ++c) {
        float vv = resid[(size_t)gr * 32 + c] + tileb[wid][lane][c];
        vals[c] = vv; m += vv;
      }
      m *= (1.f / 32.f);
      float var = 0.f;
#pragma unroll
      for (int c = 0; c < 32; ++c) { float d = vals[c] - m; var += d * d; }
      var *= (1.f / 32.f);
      float rs = rsqrtf(var + 1e-5f);
#pragma unroll
      for (int c = 0; c < 32; ++c)
        Y[(size_t)gr * 32 + c] = (vals[c] - m) * rs * gamma[c] + beta[c];
    }
  }
}

// ===========================================================================
// Y = LayerNorm(X + relu(X@W1+b1)@W2+b2)   32 -> 64 -> 32
// ===========================================================================
__global__ void ffnln_kernel(int rows, const float* __restrict__ X,
                             const __bf16* __restrict__ W1p, const float* __restrict__ b1,
                             const __bf16* __restrict__ W2p, const float* __restrict__ b2,
                             const float* __restrict__ gamma, const float* __restrict__ beta,
                             float* __restrict__ Y) {
  __shared__ float hid[WPB][16][64];
  __shared__ float outb[WPB][16][32];
  int lane = threadIdx.x & 31, wid = threadIdx.x >> 5;
  int half = lane >> 4, lr = lane & 15;
  int tiles = rows >> 4;
  __builtin_prefetch(W1p, 0, 1);
  __builtin_prefetch(W2p, 0, 1);
  for (int tile = blockIdx.x * WPB + wid; tile < tiles; tile += gridDim.x * WPB) {
    v16bf a = packA(X + (size_t)(tile * 16 + lr) * 32, 0, half);
#pragma unroll
    for (int nt = 0; nt < 4; ++nt) {
      v8f acc = {};
      acc = wmma_bf16(a, loadBp(W1p, 1, 0, nt, lane), acc);
      int col = nt * 16 + lr; float bb = b1[col];
#pragma unroll
      for (int i2 = 0; i2 < 8; ++i2) {
        float vv = acc[i2] + bb;
        hid[wid][i2 + half * 8][col] = vv > 0.f ? vv : 0.f;
      }
    }
    v16bf a2[2];
#pragma unroll
    for (int c = 0; c < 2; ++c) a2[c] = packA(&hid[wid][lr][0], c * 32, half);
#pragma unroll
    for (int nt = 0; nt < 2; ++nt) {
      v8f acc = {};
#pragma unroll
      for (int c = 0; c < 2; ++c)
        acc = wmma_bf16(a2[c], loadBp(W2p, 2, c, nt, lane), acc);
      int col = nt * 16 + lr; float bb = b2[col];
#pragma unroll
      for (int i2 = 0; i2 < 8; ++i2) outb[wid][i2 + half * 8][col] = acc[i2] + bb;
    }
    if (lane < 16) {
      int gr = tile * 16 + lane;
      float vals[32]; float m = 0.f;
#pragma unroll
      for (int c = 0; c < 32; ++c) {
        float vv = X[(size_t)gr * 32 + c] + outb[wid][lane][c];
        vals[c] = vv; m += vv;
      }
      m *= (1.f / 32.f);
      float var = 0.f;
#pragma unroll
      for (int c = 0; c < 32; ++c) { float d = vals[c] - m; var += d * d; }
      var *= (1.f / 32.f);
      float rs = rsqrtf(var + 1e-5f);
#pragma unroll
      for (int c = 0; c < 32; ++c)
        Y[(size_t)gr * 32 + c] = (vals[c] - m) * rs * gamma[c] + beta[c];
    }
  }
}

// ===========================================================================
// Segment-softmax attention over edges (batched over T via node id t*N+n)
// ===========================================================================
__global__ void attn_init_kernel(unsigned* __restrict__ smaxu, float* __restrict__ denom,
                                 float* __restrict__ agg, int n4, int n32) {
  int i = blockIdx.x * blockDim.x + threadIdx.x;
  if (i < n32) agg[i] = 0.f;
  if (i < n4) { smaxu[i] = 0u; denom[i] = 0.f; }
}

__global__ void attn_score_kernel(const float* __restrict__ q, const float* __restrict__ k,
                                  const float* __restrict__ ep, const int* __restrict__ ei,
                                  float* __restrict__ score, float* __restrict__ sE,
                                  unsigned* __restrict__ smaxu) {
  const int Ec = 12000, Nc = 3000;
  int g = blockIdx.x * blockDim.x + threadIdx.x;
  if (g >= 12 * Ec) return;
  int t = g / Ec, e = g - t * Ec;
  int src = ei[(size_t)t * 2 * Ec + e];
  int dst = ei[(size_t)t * 2 * Ec + Ec + e];
  size_t qr = ((size_t)t * Nc + dst) * 32, kr = ((size_t)t * Nc + src) * 32;
  size_t er = (size_t)g * 32;
  const float rs = 0.35355339059327373f;  // 1/sqrt(DH=8)
  float s[4] = {0.f, 0.f, 0.f, 0.f};
#pragma unroll
  for (int c = 0; c < 32; ++c) {
    float sc = q[qr + c] * k[kr + c] * rs * ep[er + c];
    score[er + c] = sc;
    s[c >> 3] += sc;
  }
  size_t dr = ((size_t)t * Nc + dst) * 4;
#pragma unroll
  for (int h = 0; h < 4; ++h) {
    sE[(size_t)g * 4 + h] = s[h];
    atomicMax(&smaxu[dr + h], fmapu(s[h]));
  }
}

__global__ void attn_accum_kernel(const float* __restrict__ v, const int* __restrict__ ei,
                                  const float* __restrict__ sE, const unsigned* __restrict__ smaxu,
                                  float* __restrict__ denom, float* __restrict__ agg) {
  const int Ec = 12000, Nc = 3000;
  int g = blockIdx.x * blockDim.x + threadIdx.x;
  if (g >= 12 * Ec) return;
  int t = g / Ec, e = g - t * Ec;
  int src = ei[(size_t)t * 2 * Ec + e];
  int dst = ei[(size_t)t * 2 * Ec + Ec + e];
  size_t dr4 = ((size_t)t * Nc + dst) * 4, dr32 = ((size_t)t * Nc + dst) * 32;
  size_t sr32 = ((size_t)t * Nc + src) * 32;
#pragma unroll
  for (int h = 0; h < 4; ++h) {
    float w = expf(sE[(size_t)g * 4 + h] - fumap(smaxu[dr4 + h]));
    atomicAdd(&denom[dr4 + h], w);
#pragma unroll
    for (int d = 0; d < 8; ++d)
      atomicAdd(&agg[dr32 + h * 8 + d], w * v[sr32 + h * 8 + d]);
  }
}

__global__ void attn_final_kernel(const float* __restrict__ agg, const float* __restrict__ denom,
                                  float* __restrict__ hattn, int n32) {
  int i = blockIdx.x * blockDim.x + threadIdx.x;
  if (i >= n32) return;
  int row = i >> 5, c = i & 31;
  hattn[i] = agg[i] / (denom[(size_t)row * 4 + (c >> 3)] + 1e-6f);
}

// ===========================================================================
// Fused dilated inception (k=2,3,6,7, dilation 2) + tanh*sigmoid gate + 1x1 adjust
// seq: [T=12][R][32] -> out: [12][R][32]
// ===========================================================================
struct TempW {
  const float* fw[4]; const float* fb[4];
  const float* gw[4]; const float* gb[4];
  const float* aw; const float* ab;
};

__global__ void temporal_kernel(const float* __restrict__ seq, float* __restrict__ out,
                                int R, TempW W) {
  const int KS[4] = {2, 3, 6, 7};
  size_t idx = blockIdx.x * (size_t)blockDim.x + threadIdx.x;
  size_t total = (size_t)12 * R * 32;
  if (idx >= total) return;
  int d = (int)(idx & 31);
  size_t rt = idx >> 5;
  int r = (int)(rt % R);
  int t = (int)(rt / R);
  float win[13];
#pragma unroll
  for (int j = 0; j < 13; ++j) {
    int tp = t - 6 + j;
    win[j] = (tp >= 0 && tp < 12) ? seq[((size_t)tp * R + r) * 32 + d] : 0.f;
  }
  float acc = W.ab[0];
  int ch = 0;
#pragma unroll
  for (int gset = 0; gset < 4; ++gset) {
    int k = KS[gset], pad = k - 1;  // (k-1)*DIL/2 with DIL=2
    const float* fw = W.fw[gset]; const float* fb = W.fb[gset];
    const float* gw = W.gw[gset]; const float* gb = W.gb[gset];
    for (int c = 0; c < 8; ++c) {
      float f = fb[c], g = gb[c];
      for (int j = 0; j < k; ++j) {
        float vv = win[6 + j * 2 - pad];
        f += fw[c * k + j] * vv;
        g += gw[c * k + j] * vv;
      }
      float gate = tanhf(f) * (1.f / (1.f + expf(-g)));
      acc += W.aw[ch] * gate;
      ++ch;
    }
  }
  out[((size_t)t * R + r) * 32 + d] = acc;
}

// ===========================================================================
// Fused classifier: feat=[dn[src]|dn[dst]|de] (96) -> relu(@W1 96x128) -> @W2 128x8
// Feature gather via GLOBAL_LOAD_ASYNC_TO_LDS_B128 (E multiple of 16 => each
// 16-edge tile lies in one timestep). Writes d_out [T*E, 8] directly.
// ===========================================================================
__global__ void clf_kernel(const float* __restrict__ dn, const float* __restrict__ de,
                           const int* __restrict__ ei,
                           const __bf16* __restrict__ W1p, const float* __restrict__ b1,
                           const __bf16* __restrict__ W2p, const float* __restrict__ b2,
                           float* __restrict__ out) {
  __shared__ __align__(16) float feat[WPB][16][96];
  __shared__ __align__(16) float hid[WPB][16][128];
  const int Ec = 12000, Nc = 3000;
  int lane = threadIdx.x & 31, wid = threadIdx.x >> 5;
  int half = lane >> 4, lr = lane & 15;
  int tiles = (12 * Ec) >> 4;
  __builtin_prefetch(W1p, 0, 1);
  __builtin_prefetch(W2p, 0, 1);
  for (int tile = blockIdx.x * WPB + wid; tile < tiles; tile += gridDim.x * WPB) {
    int t = (tile * 16) / Ec;           // tile-uniform (Ec % 16 == 0)
    int e0 = tile * 16 - t * Ec;
    // 16 rows x 96 floats = 384 aligned 16-byte chunks; 32 lanes x 12 iters.
#pragma unroll
    for (int it = 0; it < 12; ++it) {
      int c = it * 32 + lane;
      int rr = c / 24, s16 = c - rr * 24;   // 24 chunks per 96-float row
      int cc = s16 * 4;
      const float* src;
      if (cc < 32) {
        int sidx = ei[(size_t)t * 2 * Ec + (e0 + rr)];
        src = dn + ((size_t)t * Nc + sidx) * 32 + cc;
      } else if (cc < 64) {
        int didx = ei[(size_t)t * 2 * Ec + Ec + (e0 + rr)];
        src = dn + ((size_t)t * Nc + didx) * 32 + (cc - 32);
      } else {
        src = de + ((size_t)t * Ec + (e0 + rr)) * 32 + (cc - 64);
      }
      async_g2l_b128(&feat[wid][rr][cc], src);
    }
    wait_async0();
    v16bf a[3];
#pragma unroll
    for (int c = 0; c < 3; ++c) a[c] = packA(&feat[wid][lr][0], c * 32, half);
    for (int nt = 0; nt < 8; ++nt) {
      v8f acc = {};
#pragma unroll
      for (int c = 0; c < 3; ++c)
        acc = wmma_bf16(a[c], loadBp(W1p, 3, c, nt, lane), acc);
      int col = nt * 16 + lr; float bb = b1[col];
#pragma unroll
      for (int i2 = 0; i2 < 8; ++i2) {
        float vv = acc[i2] + bb;
        hid[wid][i2 + half * 8][col] = vv > 0.f ? vv : 0.f;
      }
    }
    v16bf a2[4];
#pragma unroll
    for (int c = 0; c < 4; ++c) a2[c] = packA(&hid[wid][lr][0], c * 32, half);
    v8f acc = {};
#pragma unroll
    for (int c = 0; c < 4; ++c)
      acc = wmma_bf16(a2[c], loadBp(W2p, 4, c, 0, lane), acc);
    if (lr < 8) {                       // Nout=8: lanes lr>=8 computed padded cols
      float bb = b2[lr];
      float* yp = out + (size_t)(tile * 16 + half * 8) * 8 + lr;
#pragma unroll
      for (int i2 = 0; i2 < 8; ++i2) yp[(size_t)i2 * 8] = acc[i2] + bb;
    }
  }
}

// ===========================================================================
// Host launcher
// ===========================================================================
extern "C" void kernel_launch(void* const* d_in, const int* in_sizes, int n_in,
                              void* d_out, int out_size, void* d_ws, size_t ws_size,
                              hipStream_t stream) {
  (void)in_sizes; (void)n_in; (void)out_size; (void)ws_size;
  const int T = 12, N = 3000, E = 12000;
  const int SN = T * N, SE2 = T * E;

  // ---- unpack inputs (setup_inputs() insertion order, leaves depth-first) ----
  int p = 0;
  auto nf = [&]() { return (const float*)d_in[p++]; };
  const float* x = nf();
  const int* ei = (const int*)d_in[p++];
  const float* ea = nf();
  const float *Wn1 = nf(), *bn1 = nf(), *Wn2 = nf(), *bn2 = nf();
  const float *Wei1 = nf(), *bei1 = nf(), *Wei2 = nf(), *bei2 = nf();
  struct GTP {
    const float *Wq, *bq, *Wk, *bk, *Wv, *bv, *We, *be, *Oh, *bOh, *Oe, *bOe;
    const float *Wh1, *bh1, *Wh2, *bh2, *We1, *be1, *We2, *be2;
    const float *g1h, *t1h, *g2h, *t2h, *g1e, *t1e, *g2e, *t2e;
  } gt[2];
  for (int l = 0; l < 2; ++l) {
    GTP& g = gt[l];
    g.Wq = nf(); g.bq = nf(); g.Wk = nf(); g.bk = nf(); g.Wv = nf(); g.bv = nf();
    g.We = nf(); g.be = nf(); g.Oh = nf(); g.bOh = nf(); g.Oe = nf(); g.bOe = nf();
    g.Wh1 = nf(); g.bh1 = nf(); g.Wh2 = nf(); g.bh2 = nf();
    g.We1 = nf(); g.be1 = nf(); g.We2 = nf(); g.be2 = nf();
    g.g1h = nf(); g.t1h = nf(); g.g2h = nf(); g.t2h = nf();
    g.g1e = nf(); g.t1e = nf(); g.g2e = nf(); g.t2e = nf();
  }
  const float *Wef1 = nf(), *bef1 = nf(), *Wef2 = nf(), *bef2 = nf();
  TempW tn, te;
  for (int j = 0; j < 4; ++j) { tn.fw[j] = nf(); tn.fb[j] = nf(); }
  for (int j = 0; j < 4; ++j) { tn.gw[j] = nf(); tn.gb[j] = nf(); }
  for (int j = 0; j < 4; ++j) { te.fw[j] = nf(); te.fb[j] = nf(); }
  for (int j = 0; j < 4; ++j) { te.gw[j] = nf(); te.gb[j] = nf(); }
  tn.aw = nf(); tn.ab = nf();
  te.aw = nf(); te.ab = nf();
  const float *C1w = nf(), *C1b = nf(), *C2w = nf(), *C2b = nf();

  // ---- workspace arena (~134 MB) ----
  float* ws = (float*)d_ws;
  float* h_seq = ws;                            // SN*32
  float* e_seq = h_seq + (size_t)SN * 32;       // SE2*32
  float* qb    = e_seq + (size_t)SE2 * 32;      // SN*32
  float* kb    = qb + (size_t)SN * 32;          // SN*32
  float* vb    = kb + (size_t)SN * 32;          // SN*32
  float* epb   = vb + (size_t)SN * 32;          // SE2*32
  float* score = epb + (size_t)SE2 * 32;        // SE2*32
  float* sEb   = score + (size_t)SE2 * 32;      // SE2*4
  unsigned* smaxu = (unsigned*)(sEb + (size_t)SE2 * 4);  // SN*4
  float* denom = (float*)(smaxu + (size_t)SN * 4);       // SN*4
  float* agg   = denom + (size_t)SN * 4;        // SN*32
  float* hattn = agg + (size_t)SN * 32;         // SN*32
  float* h1b   = hattn + (size_t)SN * 32;       // SN*32
  float* e1b   = h1b + (size_t)SN * 32;         // SE2*32
  float* dn    = e1b + (size_t)SE2 * 32;        // SN*32
  float* de    = dn + (size_t)SN * 32;          // SE2*32
  __bf16* pk   = (__bf16*)(de + (size_t)SE2 * 32);  // packed weights (~128 KB)

  // pack a weight matrix into WMMA B layout; returns packed pointer
  auto packw = [&](const float* W, int K, int Nout) -> const __bf16* {
    int KC = K / 32, NT = (Nout + 15) / 16;
    int total = KC * NT * 32;
    __bf16* dst = pk;
    packW_kernel<<<(total + 127) / 128, 128, 0, stream>>>(W, K, Nout, dst);
    pk += (size_t)total * 16;
    return dst;
  };

  const __bf16* pWn1 = packw(Wn1, 64, 32);
  const __bf16* pWn2 = packw(Wn2, 32, 32);
  const __bf16* pWei1 = packw(Wei1, 32, 32);
  const __bf16* pWei2 = packw(Wei2, 32, 32);
  const __bf16 *pQ[2], *pK[2], *pV[2], *pWe[2], *pOh[2], *pOe[2];
  const __bf16 *pH1[2], *pH2[2], *pE1[2], *pE2[2];
  for (int l = 0; l < 2; ++l) {
    pQ[l] = packw(gt[l].Wq, 32, 32);  pK[l] = packw(gt[l].Wk, 32, 32);
    pV[l] = packw(gt[l].Wv, 32, 32);  pWe[l] = packw(gt[l].We, 32, 32);
    pOh[l] = packw(gt[l].Oh, 32, 32); pOe[l] = packw(gt[l].Oe, 32, 32);
    pH1[l] = packw(gt[l].Wh1, 32, 64); pH2[l] = packw(gt[l].Wh2, 64, 32);
    pE1[l] = packw(gt[l].We1, 32, 64); pE2[l] = packw(gt[l].We2, 64, 32);
  }
  const __bf16* pWef1 = packw(Wef1, 32, 32);
  const __bf16* pWef2 = packw(Wef2, 32, 32);
  const __bf16* pC1 = packw(C1w, 96, 128);
  const __bf16* pC2 = packw(C2w, 128, 8);

  auto gb = [](int rows) { int tiles = rows / 16; return (tiles + WPB - 1) / WPB; };

  // 1. input MLPs (batched over T)
  mlp2_kernel<64><<<gb(SN), 128, 0, stream>>>(SN, x, pWn1, bn1, pWn2, bn2, h_seq);
  mlp2_kernel<32><<<gb(SE2), 128, 0, stream>>>(SE2, ea, pWei1, bei1, pWei2, bei2, e_seq);

  // 2. two graph-transformer layers
  for (int l = 0; l < 2; ++l) {
    const GTP& g = gt[l];
    gemm32_kernel<<<gb(SN), 128, 0, stream>>>(SN, h_seq, pQ[l], g.bq, qb);
    gemm32_kernel<<<gb(SN), 128, 0, stream>>>(SN, h_seq, pK[l], g.bk, kb);
    gemm32_kernel<<<gb(SN), 128, 0, stream>>>(SN, h_seq, pV[l], g.bv, vb);
    gemm32_kernel<<<gb(SE2), 128, 0, stream>>>(SE2, e_seq, pWe[l], g.be, epb);
    attn_init_kernel<<<(SN * 32 + 255) / 256, 256, 0, stream>>>(smaxu, denom, agg, SN * 4, SN * 32);
    attn_score_kernel<<<(SE2 + 255) / 256, 256, 0, stream>>>(qb, kb, epb, ei, score, sEb, smaxu);
    attn_accum_kernel<<<(SE2 + 255) / 256, 256, 0, stream>>>(vb, ei, sEb, smaxu, denom, agg);
    attn_final_kernel<<<(SN * 32 + 255) / 256, 256, 0, stream>>>(agg, denom, hattn, SN * 32);
    oln_kernel<<<gb(SN), 128, 0, stream>>>(SN, hattn, pOh[l], g.bOh, h_seq, g.g1h, g.t1h, h1b);
    ffnln_kernel<<<gb(SN), 128, 0, stream>>>(SN, h1b, pH1[l], g.bh1, pH2[l], g.bh2, g.g2h, g.t2h, h_seq);
    oln_kernel<<<gb(SE2), 128, 0, stream>>>(SE2, score, pOe[l], g.bOe, e_seq, g.g1e, g.t1e, e1b);
    ffnln_kernel<<<gb(SE2), 128, 0, stream>>>(SE2, e1b, pE1[l], g.be1, pE2[l], g.be2, g.g2e, g.t2e, e_seq);
  }

  // 3. edge output MLP
  mlp2_kernel<32><<<gb(SE2), 128, 0, stream>>>(SE2, e_seq, pWef1, bef1, pWef2, bef2, e1b);

  // 4. temporal dilated inception + gating + 1x1 channel adjust
  temporal_kernel<<<(int)(((size_t)T * N * 32 + 255) / 256), 256, 0, stream>>>(h_seq, dn, N, tn);
  temporal_kernel<<<(int)(((size_t)T * E * 32 + 255) / 256), 256, 0, stream>>>(e1b, de, E, te);

  // 5. fused classifier -> d_out [T,E,8]
  clf_kernel<<<gb(SE2), 128, 0, stream>>>(dn, de, ei, pC1, C1b, pC2, C2b, (float*)d_out);
}
